// MobileNetV2ME_15083925143722
// MI455X (gfx1250) — matrix operand order repaired and checked by reading, not compiled, not run on hardware
//
#include <hip/hip_runtime.h>

// MobileNetV2-ME (sparse event net) for gfx1250.
// 1x1 convs = WMMA f16 GEMMs (v_wmma_f32_16x16x32_f16). Cin and the N-blocking
// factor NB are compile-time: the K-loop fully unrolls and each wave computes
// a 16 x (16*NB) tile, loading A once per K-chunk and issuing NB independent
// WMMAs (NB x less A traffic; this workload is L2-bandwidth bound, K is small).
// Weights are pre-packed into the exact WMMA B lane layout (f16, tail-padded)
// so the inner loop is pure b128 loads + wmma. BN/relu6/mask/residual fused
// into epilogues; conv8 (32->1280) fused with the masked global pool so its
// 84MB output never touches memory. Activations are f16 -> working set stays
// L2-resident (192MB).

typedef __attribute__((ext_vector_type(16))) _Float16 v16h;
typedef __attribute__((ext_vector_type(8)))  _Float16 v8h;
typedef __attribute__((ext_vector_type(8)))  float    v8f;

#define BN_EPS 1e-5f

__device__ __forceinline__ float relu6f(float x) { return fminf(fmaxf(x, 0.f), 6.f); }

// ---------------- utility kernels ----------------

__global__ void zero_f32_kernel(float* p, int n) {
    int i = blockIdx.x * blockDim.x + threadIdx.x;
    if (i < n) p[i] = 0.f;
}

// mask (bool bytes, 512x512) -> m0 (f32 256x256), 2x2 "any"
__global__ void mask0_kernel(const unsigned char* __restrict__ mask, float* __restrict__ m0, int B) {
    int i = blockIdx.x * blockDim.x + threadIdx.x;
    int total = B * 256 * 256;
    if (i >= total) return;
    int wo = i & 255; int t = i >> 8; int ho = t & 255; int b = t >> 8;
    const unsigned char* mp = mask + (((size_t)b * 512 + (size_t)ho * 2) * 512 + (size_t)wo * 2);
    unsigned char v = (unsigned char)(mp[0] | mp[1] | mp[512] | mp[513]);
    m0[i] = v ? 1.f : 0.f;
}

// 2x2 max downsample of a f32 mask; optionally accumulate per-batch active count
__global__ void maskdown_kernel(const float* __restrict__ mi, float* __restrict__ mo,
                                int B, int Ho, int Wo, float* cnt) {
    int i = blockIdx.x * blockDim.x + threadIdx.x;
    int total = B * Ho * Wo;
    if (i >= total) return;
    int wo = i % Wo; int t = i / Wo; int ho = t % Ho; int b = t / Ho;
    int Wi = Wo * 2;
    const float* p = mi + (((size_t)b * (Ho * 2) + (size_t)ho * 2) * Wi + (size_t)wo * 2);
    float v = fmaxf(fmaxf(p[0], p[1]), fmaxf(p[Wi], p[Wi + 1]));
    mo[i] = v;
    if (cnt && v > 0.f) atomicAdd(&cnt[b], v);
}

// ---------------- weight pack: f32 [Cin,Cout] -> f16 WMMA-B lane layout ------
// bp[((kt*ntiles + tn)*32 + lane)*16 + h] = w[K, N],
//   K = kt*32 + (lane>=16 ? 8 : 0) + (h&7) + (h>=8 ? 16 : 0),  N = tn*16 + (lane&15),
// zero-padded past Cin/Cout so the GEMM needs no guards at all.
__global__ void packw_kernel(const float* __restrict__ w, _Float16* __restrict__ bp,
                             int Cin, int Cout, int ntiles, int total) {
    int i = blockIdx.x * blockDim.x + threadIdx.x;
    if (i >= total) return;
    int h = i & 15;
    int lane = (i >> 4) & 31;
    int t = i >> 9;
    int tn = t % ntiles;
    int kt = t / ntiles;
    int K = kt * 32 + ((lane >> 4) << 3) + (h & 7) + ((h >> 3) << 4);
    int N = tn * 16 + (lane & 15);
    bp[i] = (K < Cin && N < Cout) ? (_Float16)w[(size_t)K * Cout + N] : (_Float16)0.f;
}

// ---------------- stem: 3x3 s2 conv 2->32 + BN + relu6 + mask ----------------
// SAME pad for 512->256 s2 k3: pad_before = 0. Input is x * mask (per-site).
__global__ void stem_kernel(const float* __restrict__ x, const unsigned char* __restrict__ mask,
                            const float* __restrict__ w,
                            const float* __restrict__ g, const float* __restrict__ bb,
                            const float* __restrict__ mm, const float* __restrict__ vv,
                            const float* __restrict__ m0, _Float16* __restrict__ out, int B) {
    int i = blockIdx.x * blockDim.x + threadIdx.x;
    int total = B * 256 * 256 * 32;
    if (i >= total) return;
    int co = i & 31; int t = i >> 5;
    int wo = t & 255; t >>= 8; int ho = t & 255; int b = t >> 8;
    float acc = 0.f;
    #pragma unroll
    for (int dy = 0; dy < 3; ++dy) {
        int yi = ho * 2 + dy;
        if (yi >= 512) continue;
        #pragma unroll
        for (int dx = 0; dx < 3; ++dx) {
            int xi = wo * 2 + dx;
            if (xi >= 512) continue;
            size_t si = ((size_t)b * 512 + yi) * 512 + xi;
            if (mask[si]) {
                const float* xp = x + si * 2;
                const float* wp = w + (size_t)((dy * 3 + dx) * 2) * 32 + co;
                acc += xp[0] * wp[0] + xp[1] * wp[32];
            }
        }
    }
    float sc = g[co] * rsqrtf(vv[co] + BN_EPS);
    float sh = bb[co] - mm[co] * sc;
    float y = relu6f(acc * sc + sh) * m0[((size_t)b * 256 + ho) * 256 + wo];
    out[i] = (_Float16)y;
}

// ---------------- depthwise 3x3 (+BN+relu6+mask) ----------------
__global__ void dw_kernel(const _Float16* __restrict__ in, const float* __restrict__ w,
                          const float* __restrict__ g, const float* __restrict__ bb,
                          const float* __restrict__ mm, const float* __restrict__ vv,
                          const float* __restrict__ mo, _Float16* __restrict__ out,
                          int B, int Hi, int Wi, int Ho, int Wo, int C, int s, int pad) {
    int i = blockIdx.x * blockDim.x + threadIdx.x;
    int total = B * Ho * Wo * C;
    if (i >= total) return;
    int c = i % C; int t = i / C;
    int wo = t % Wo; t /= Wo; int ho = t % Ho; int b = t / Ho;
    float acc = 0.f;
    #pragma unroll
    for (int dy = 0; dy < 3; ++dy) {
        int yi = ho * s + dy - pad;
        if ((unsigned)yi >= (unsigned)Hi) continue;
        #pragma unroll
        for (int dx = 0; dx < 3; ++dx) {
            int xi = wo * s + dx - pad;
            if ((unsigned)xi >= (unsigned)Wi) continue;
            acc += (float)in[(((size_t)b * Hi + yi) * Wi + xi) * C + c] * w[(dy * 3 + dx) * C + c];
        }
    }
    float sc = g[c] * rsqrtf(vv[c] + BN_EPS);
    float sh = bb[c] - mm[c] * sc;
    float y = relu6f(acc * sc + sh) * mo[((size_t)b * Ho + ho) * Wo + wo];
    out[i] = (_Float16)y;
}

// ---------------- 1x1 conv GEMM via WMMA, Cin + N-block compile-time ---------
// in: f16 [P, CIN]; bp: packed f16 B tiles; out: f16 [P, Cout].
// One wave = 16(pixels) x 16*NB(channels); A loaded once per K-chunk, NB
// independent WMMA accumulator chains. Host guarantees NB | ceil(Cout/16).
template <int CIN, int NB>
__global__ __launch_bounds__(128)
void gemm1x1_t(const _Float16* __restrict__ in, const _Float16* __restrict__ bp,
               const float* __restrict__ g, const float* __restrict__ bb,
               const float* __restrict__ mm, const float* __restrict__ vv,
               const float* __restrict__ maskp, const _Float16* __restrict__ resid,
               _Float16* __restrict__ out, int P, int Cout, int relu) {
    constexpr int KT = (CIN + 31) / 32;
    int ntiles = (Cout + 15) >> 4;
    int ngroups = ntiles / NB;
    int waves = (P >> 4) * ngroups;
    int wave = (blockIdx.x * blockDim.x + threadIdx.x) >> 5;
    if (wave >= waves) return;              // wave-uniform: EXEC all-ones for WMMA
    int gn = wave % ngroups, tm = wave / ngroups;
    int lane = threadIdx.x & 31;
    int l15 = lane & 15;
    int hi = lane >> 4;
    size_t rowA = (size_t)(tm * 16 + l15) * (size_t)CIN;
    v8f acc[NB];
    #pragma unroll
    for (int j = 0; j < NB; ++j) { v8f z = {}; acc[j] = z; }
    #pragma unroll
    for (int kt = 0; kt < KT; ++kt) {
        const int k0 = kt * 32;
        // A tile (ISA layout): halves 0..7 -> K=k0+8*hi.., halves 8..15 -> +16.
        // Guards constant-fold for every CIN except the 24 boundary chunk.
        v8h r0 = {0, 0, 0, 0, 0, 0, 0, 0};
        v8h r1 = {0, 0, 0, 0, 0, 0, 0, 0};
        if (k0 + hi * 8 < CIN)      r0 = *reinterpret_cast<const v8h*>(in + rowA + k0 + hi * 8);
        if (k0 + hi * 8 + 16 < CIN) r1 = *reinterpret_cast<const v8h*>(in + rowA + k0 + hi * 8 + 16);
        v16h a;
        #pragma unroll
        for (int j = 0; j < 8; ++j) { a[j] = r0[j]; a[j + 8] = r1[j]; }
        // NB B tiles: 32 consecutive bytes per lane each, reusing A.
        #pragma unroll
        for (int j = 0; j < NB; ++j) {
            const _Float16* bptr = bp + (((size_t)kt * ntiles + gn * NB + j) * 32 + lane) * 16;
            v8h b0 = *reinterpret_cast<const v8h*>(bptr);
            v8h b1 = *reinterpret_cast<const v8h*>(bptr + 8);
            v16h bm;
            #pragma unroll
            for (int q = 0; q < 8; ++q) { bm[q] = b0[q]; bm[q + 8] = b1[q]; }
            acc[j] = __builtin_amdgcn_wmma_f32_16x16x32_f16(false, a, false, bm,
                                                            (short)0, acc[j], false, false);
        }
    }
    // epilogue: lane owns one channel per N-tile, 8 pixels (C/D layout).
    int pix0 = tm * 16 + (hi << 3);
    float mk[8];
    #pragma unroll
    for (int r = 0; r < 8; ++r) mk[r] = maskp[pix0 + r];
    #pragma unroll
    for (int j = 0; j < NB; ++j) {
        int co = (gn * NB + j) * 16 + l15;
        if (co < Cout) {
            float sc = g[co] * rsqrtf(vv[co] + BN_EPS);
            float sh = bb[co] - mm[co] * sc;
            #pragma unroll
            for (int r = 0; r < 8; ++r) {
                int pix = pix0 + r;
                float y = acc[j][r] * sc + sh;
                if (relu) y = relu6f(y);
                y *= mk[r];
                if (resid) y += (float)resid[(size_t)pix * Cout + co];
                out[(size_t)pix * Cout + co] = (_Float16)y;
            }
        }
    }
}

// ---------------- conv8 (32->1280) fused with masked global pooling ----------
// Cin=32 (KT=1), NB=4: one A load feeds 4 WMMAs; epilogue reduces 8 pixels per
// lane per tile, one f32 atomic each. Tiles never straddle batches (4096%16==0).
__global__ __launch_bounds__(128)
void gemm_pool_kernel(const _Float16* __restrict__ in, const _Float16* __restrict__ bp,
                      const float* __restrict__ g, const float* __restrict__ bb,
                      const float* __restrict__ mm, const float* __restrict__ vv,
                      const float* __restrict__ maskp, float* __restrict__ pooled,
                      int P, int Cout, int HW) {
    constexpr int NB = 4;
    int ntiles = Cout >> 4;                 // 1280 -> 80
    int ngroups = ntiles / NB;              // 20
    int waves = (P >> 4) * ngroups;
    int wave = (blockIdx.x * blockDim.x + threadIdx.x) >> 5;
    if (wave >= waves) return;
    int gn = wave % ngroups, tm = wave / ngroups;
    int lane = threadIdx.x & 31;
    int l15 = lane & 15;
    int hi = lane >> 4;
    size_t rowA = (size_t)(tm * 16 + l15) * 32;
    v8h r0 = *reinterpret_cast<const v8h*>(in + rowA + hi * 8);
    v8h r1 = *reinterpret_cast<const v8h*>(in + rowA + hi * 8 + 16);
    v16h a;
    #pragma unroll
    for (int j = 0; j < 8; ++j) { a[j] = r0[j]; a[j + 8] = r1[j]; }
    v8f acc[NB];
    #pragma unroll
    for (int j = 0; j < NB; ++j) {
        const _Float16* bptr = bp + (((size_t)gn * NB + j) * 32 + lane) * 16;
        v8h b0 = *reinterpret_cast<const v8h*>(bptr);
        v8h b1 = *reinterpret_cast<const v8h*>(bptr + 8);
        v16h bm;
        #pragma unroll
        for (int q = 0; q < 8; ++q) { bm[q] = b0[q]; bm[q + 8] = b1[q]; }
        v8f z = {};
        acc[j] = __builtin_amdgcn_wmma_f32_16x16x32_f16(false, a, false, bm,
                                                        (short)0, z, false, false);
    }
    int pix0 = tm * 16 + (hi << 3);
    int b = pix0 / HW;
    float mk[8];
    #pragma unroll
    for (int r = 0; r < 8; ++r) mk[r] = maskp[pix0 + r];
    #pragma unroll
    for (int j = 0; j < NB; ++j) {
        int co = (gn * NB + j) * 16 + l15;
        float sc = g[co] * rsqrtf(vv[co] + BN_EPS);
        float sh = bb[co] - mm[co] * sc;
        float s = 0.f;
        #pragma unroll
        for (int r = 0; r < 8; ++r) {
            s += relu6f(acc[j][r] * sc + sh) * mk[r];
        }
        atomicAdd(&pooled[(size_t)b * Cout + co], s);
    }
}

// ---------------- final FC: out[b,k] = (pooled[b,:]/cnt[b]) @ fc_w + fc_b -----
__global__ void fc_kernel(const float* __restrict__ pooled, const float* __restrict__ cnt,
                          const float* __restrict__ fw, const float* __restrict__ fb,
                          float* __restrict__ out, int B) {
    int i = blockIdx.x * blockDim.x + threadIdx.x;
    if (i >= B * 10) return;
    int k = i % 10, b = i / 10;
    float c = fmaxf(cnt[b], 1.f);
    float s = 0.f;
    for (int j = 0; j < 1280; ++j) s += pooled[(size_t)b * 1280 + j] * fw[j * 10 + k];
    out[i] = s / c + fb[k];
}

// ---------------- host launcher ----------------

struct BlkCfg { int cin, ch, cout, s, res, nbe, nbp; };

extern "C" void kernel_launch(void* const* d_in, const int* in_sizes, int n_in,
                              void* d_out, int out_size, void* d_ws, size_t ws_size,
                              hipStream_t stream) {
    (void)in_sizes; (void)n_in; (void)out_size;
    // Input order (setup_inputs insertion order; nested dicts in insertion order,
    // bn dicts as {g,b,m,v}):
    //  0: x  1: mask  2: stem_w  3-6: stem_bn{g,b,m,v}
    //  7 + 15*i: block i = { w1, bn1{g,b,m,v}, w2, bn2{g,b,m,v}, w3, bn3{g,b,m,v} }
    //  97: w8  98-101: bn8{g,b,m,v}  102: fc_w  103: fc_b
    const float*         X    = (const float*)d_in[0];
    const unsigned char* MASK = (const unsigned char*)d_in[1];
    const float*         STW  = (const float*)d_in[2];
    const float* STG = (const float*)d_in[3];
    const float* STB = (const float*)d_in[4];
    const float* STM = (const float*)d_in[5];
    const float* STV = (const float*)d_in[6];
    const float* W8  = (const float*)d_in[97];
    const float* B8G = (const float*)d_in[98];
    const float* B8B = (const float*)d_in[99];
    const float* B8M = (const float*)d_in[100];
    const float* B8V = (const float*)d_in[101];
    const float* FCW = (const float*)d_in[102];
    const float* FCB = (const float*)d_in[103];

    // ---- workspace layout (f16 activations; ~208MB total) ----
    size_t off = 0;
    auto carve = [&](size_t bytes) -> char* {
        char* p = (char*)d_ws + off;
        off += (bytes + 255) & ~(size_t)255;
        return p;
    };
    float*     m0     = (float*)carve((size_t)8 * 256 * 256 * 4);
    float*     m1     = (float*)carve((size_t)8 * 128 * 128 * 4);
    float*     m2     = (float*)carve((size_t)8 * 64 * 64 * 4);
    float*     pooled = (float*)carve((size_t)(8 * 1280 + 8) * 4);  // + cnt[8] tail
    float*     cnt    = pooled + 8 * 1280;
    _Float16*  WPK    = (_Float16*)carve((size_t)131072 * 2);       // packed weights (~185KB used)
    _Float16*  Hb     = (_Float16*)carve((size_t)16777216 * 2);     // 8*256*256*32
    _Float16*  H2b    = (_Float16*)carve((size_t)16777216 * 2);
    _Float16*  Tb     = (_Float16*)carve((size_t)50331648 * 2);     // 8*256*256*96 max expand
    _Float16*  Db     = (_Float16*)carve((size_t)18874368 * 2);     // 8*128*128*144 max dw out
    if (off > ws_size) return;  // workspace too small; bail deterministically

    auto cdiv = [](long a, long b) { return (int)((a + b - 1) / b); };
    float* ms[3] = { m0, m1, m2 };

    // ---- pack all 1x1 weights into WMMA B layout (tiny, L2-resident) ----
    size_t wpk_off = 0;
    auto pack = [&](const float* w, int Cin, int Cout) -> _Float16* {
        int kt = (Cin + 31) / 32, nt = (Cout + 15) / 16;
        int total = kt * nt * 512;
        _Float16* dst = WPK + wpk_off;
        wpk_off += (size_t)total;
        packw_kernel<<<cdiv(total, 256), 256, 0, stream>>>(w, dst, Cin, Cout, nt, total);
        return dst;
    };

    // masks + counts (re-zeroed every call: atomics accumulate)
    zero_f32_kernel<<<cdiv(8 * 1280 + 8, 256), 256, 0, stream>>>(pooled, 8 * 1280 + 8);
    mask0_kernel<<<cdiv((long)8 * 256 * 256, 256), 256, 0, stream>>>(MASK, m0, 8);
    maskdown_kernel<<<cdiv((long)8 * 128 * 128, 256), 256, 0, stream>>>(m0, m1, 8, 128, 128, nullptr);
    maskdown_kernel<<<cdiv((long)8 * 64 * 64, 256), 256, 0, stream>>>(m1, m2, 8, 64, 64, cnt);

    // stem
    stem_kernel<<<cdiv((long)8 * 256 * 256 * 32, 256), 256, 0, stream>>>(
        X, MASK, STW, STG, STB, STM, STV, m0, Hb, 8);

    // dispatch over the (CIN, NB) pairs actually used by the network
    auto launch_gemm = [&](const _Float16* in, const _Float16* bp, const float* const* bn,
                           const float* mk, const _Float16* res, _Float16* out,
                           int P, int Cin, int Cout, int relu, int NB) {
        int ntiles = (Cout + 15) >> 4;
        int blocks = cdiv((long)(P >> 4) * (ntiles / NB), 4);
        #define LG(CI, NBv) gemm1x1_t<CI, NBv><<<blocks, 128, 0, stream>>>( \
            in, bp, bn[0], bn[1], bn[2], bn[3], mk, res, out, P, Cout, relu)
        if      (Cin == 32  && NB == 2) LG(32, 2);
        else if (Cin == 32  && NB == 1) LG(32, 1);
        else if (Cin == 32  && NB == 4) LG(32, 4);
        else if (Cin == 16  && NB == 2) LG(16, 2);
        else if (Cin == 96  && NB == 2) LG(96, 2);
        else if (Cin == 24  && NB == 3) LG(24, 3);
        else if (Cin == 144 && NB == 2) LG(144, 2);
        else if (Cin == 192 && NB == 2) LG(192, 2);
        #undef LG
    };

    // {cin, ch, cout, s, res, NB_expand, NB_project}; NB divides ceil(Cout/16)
    const BlkCfg cfg[6] = { {32, 32, 16, 1, 0, 2, 1}, {16, 96, 24, 2, 0, 2, 2},
                            {24, 144, 24, 1, 1, 3, 2}, {24, 144, 32, 2, 0, 3, 2},
                            {32, 192, 32, 1, 1, 4, 2}, {32, 192, 32, 1, 1, 4, 2} };

    _Float16* hin = Hb;
    _Float16* hout = H2b;
    int Hc = 256, lvl = 0;
    for (int i = 0; i < 6; ++i) {
        const BlkCfg c = cfg[i];
        int base = 7 + 15 * i;
        const float* w1 = (const float*)d_in[base + 0];
        const float* bn1[4] = { (const float*)d_in[base + 1], (const float*)d_in[base + 2],
                                (const float*)d_in[base + 3], (const float*)d_in[base + 4] };
        const float* w2 = (const float*)d_in[base + 5];
        const float* bn2[4] = { (const float*)d_in[base + 6], (const float*)d_in[base + 7],
                                (const float*)d_in[base + 8], (const float*)d_in[base + 9] };
        const float* w3 = (const float*)d_in[base + 10];
        const float* bn3[4] = { (const float*)d_in[base + 11], (const float*)d_in[base + 12],
                                (const float*)d_in[base + 13], (const float*)d_in[base + 14] };

        _Float16* bp1 = pack(w1, c.cin, c.ch);
        _Float16* bp3 = pack(w3, c.ch, c.cout);

        int P = 8 * Hc * Hc;
        launch_gemm(hin, bp1, bn1, ms[lvl], nullptr, Tb, P, c.cin, c.ch, 1, c.nbe);  // expand
        int Ho = Hc / c.s;
        int Po = 8 * Ho * Ho;
        int lvlo = (c.s == 2) ? lvl + 1 : lvl;
        int pad = (c.s == 1) ? 1 : 0;  // jax SAME: k3 s1 -> pad_before 1; s2 -> 0
        dw_kernel<<<cdiv((long)Po * c.ch, 256), 256, 0, stream>>>(
            Tb, w2, bn2[0], bn2[1], bn2[2], bn2[3], ms[lvlo], Db,
            8, Hc, Hc, Ho, Ho, c.ch, c.s, pad);                                      // depthwise
        launch_gemm(Db, bp3, bn3, ms[lvlo], c.res ? hin : nullptr, hout,
                    Po, c.ch, c.cout, 0, c.nbp);                                     // project
        _Float16* tmp = hin; hin = hout; hout = tmp;
        Hc = Ho; lvl = lvlo;
    }

    // conv8 (32->1280) + masked pool, fused (NB=4)
    {
        _Float16* bp8 = pack(W8, 32, 1280);
        int P = 8 * 64 * 64;
        int waves = (P >> 4) * ((1280 >> 4) / 4);
        gemm_pool_kernel<<<cdiv(waves, 4), 128, 0, stream>>>(
            hin, bp8, B8G, B8B, B8M, B8V, m2, pooled, P, 1280, 64 * 64);
    }

    // final FC -> d_out (8x10 f32)
    fc_kernel<<<1, 128, 0, stream>>>(pooled, cnt, FCW, FCB, (float*)d_out, 8);
}
